// QKVAttention_65223373357372
// MI455X (gfx1250) — matrix operand order repaired and checked by reading, not compiled, not run on hardware
//
#include <hip/hip_runtime.h>

typedef __attribute__((ext_vector_type(16))) _Float16 v16h;
typedef __attribute__((ext_vector_type(8)))  _Float16 v8h;
typedef __attribute__((ext_vector_type(4)))  _Float16 v4h;
typedef __attribute__((ext_vector_type(8)))  float    v8f;
typedef __attribute__((ext_vector_type(4)))  float    v4f;
typedef _Float16 half_t;

#define T_SEQ 2048
#define CH    128
#define NBAT  32
#define WPB   8            // waves per block; block covers 8 consecutive q-tiles of one batch
#define KT_STRIDE 136      // padded halves per key-row of K^T tile (32 x 128, +8 pad)
#define VR_STRIDE 40       // padded halves per channel-row of V tile (128 x 32, +8 pad)

// One wave owns one 16-query tile. All 8 waves of a block share the same batch
// and stream the same K/V tiles, staged in LDS as f16.
// GEMM1: S^T[key,q] = K_tile(16key x 32ch) x Q(32ch x 16q)
// GEMM2: O[ch,q]   += V_tile(16ch x 32key) x P(32key x 16q)
__global__ __launch_bounds__(256)
void qkv_attn_wmma_lds(const float* __restrict__ qkv, float* __restrict__ out) {
    __shared__ _Float16 lds_k[32 * KT_STRIDE];   // [key][ch]  (transposed K)
    __shared__ _Float16 lds_v[CH * VR_STRIDE];   // [ch][key]

    const int tid     = threadIdx.x;
    const int lane    = tid & 31;
    const int wave    = tid >> 5;
    const int tile    = blockIdx.x * WPB + wave;
    const int b       = tile >> 7;          // 128 q-tiles per batch
    const int qt0     = (tile & 127) << 4;
    const int half_id = lane >> 4;
    const int l16     = lane & 15;

    const float scale = 0.08838834764831845f; // 1/sqrt(128)

    const float* qbase = qkv + (size_t)b * 3 * CH * T_SEQ;
    const float* kbase = qbase + (size_t)CH * T_SEQ;
    const float* vbase = qbase + (size_t)(2 * CH) * T_SEQ;

    // ---- Q fragments (one-time): B-operand layout (32ch x 16q), 4 chunks.
    // lanes 0-15 hold rows 0-15 (VGPR v packs rows 2v,2v+1), lanes 16-31 rows 16-31.
    v16h qf[4];
#pragma unroll
    for (int cc = 0; cc < 4; ++cc) {
#pragma unroll
        for (int v = 0; v < 8; ++v) {
            const int c0 = cc * 32 + half_id * 16 + 2 * v;
            qf[cc][2 * v]     = (half_t)(qbase[(size_t)c0 * T_SEQ + qt0 + l16] * scale);
            qf[cc][2 * v + 1] = (half_t)(qbase[(size_t)(c0 + 1) * T_SEQ + qt0 + l16] * scale);
        }
    }

    float m_run = -3.0e38f;
    float l_run = 0.0f;
    v8f acc[8] = {};   // acc[n][v] = O[c = 16n + v + 8*half, q = l16]

    for (int kt = 0; kt < T_SEQ; kt += 32) {
        __syncthreads();   // previous iteration's LDS reads complete

        // ---- Stage K tile: global [ch][key] f32 -> LDS [key][ch] f16 (transpose).
        // 128ch x 32key = 1024 float4 quads; 4 per thread, coalesced b128 loads.
#pragma unroll
        for (int i = 0; i < 4; ++i) {
            const int quad = tid + 256 * i;
            const int ch   = quad >> 3;
            const int key0 = (quad & 7) * 4;
            const v4f x = *(const v4f*)(kbase + (size_t)ch * T_SEQ + kt + key0);
            lds_k[(key0 + 0) * KT_STRIDE + ch] = (half_t)x[0];
            lds_k[(key0 + 1) * KT_STRIDE + ch] = (half_t)x[1];
            lds_k[(key0 + 2) * KT_STRIDE + ch] = (half_t)x[2];
            lds_k[(key0 + 3) * KT_STRIDE + ch] = (half_t)x[3];
        }
        // ---- Stage V tile: global [ch][key] f32 -> LDS [ch][key] f16 (copy+convert).
#pragma unroll
        for (int i = 0; i < 4; ++i) {
            const int quad = tid + 256 * i;
            const int ch   = quad >> 3;
            const int key0 = (quad & 7) * 4;
            const v4f x = *(const v4f*)(vbase + (size_t)ch * T_SEQ + kt + key0);
            v4h h;
            h[0] = (half_t)x[0]; h[1] = (half_t)x[1];
            h[2] = (half_t)x[2]; h[3] = (half_t)x[3];
            *(v4h*)(lds_v + ch * VR_STRIDE + key0) = h;   // ds_store_b64
        }
        __syncthreads();   // tiles visible to all waves

        // ---- GEMM1: two 16-key score fragments from LDS.
        // A-operand per lane = two runs of 8 consecutive channels -> 2x ds_load_b128.
        v8f s0 = {};
        v8f s1 = {};
#pragma unroll
        for (int cc = 0; cc < 4; ++cc) {
            const _Float16* kr0 = lds_k + l16 * KT_STRIDE + cc * 32 + 16 * half_id;
            const _Float16* kr1 = lds_k + (16 + l16) * KT_STRIDE + cc * 32 + 16 * half_id;
            const v8h a0 = *(const v8h*)(kr0);
            const v8h b0 = *(const v8h*)(kr0 + 8);
            const v8h a1 = *(const v8h*)(kr1);
            const v8h b1 = *(const v8h*)(kr1 + 8);
            const v16h kf0 = __builtin_shufflevector(a0, b0, 0, 1, 2, 3, 4, 5, 6, 7,
                                                     8, 9, 10, 11, 12, 13, 14, 15);
            const v16h kf1 = __builtin_shufflevector(a1, b1, 0, 1, 2, 3, 4, 5, 6, 7,
                                                     8, 9, 10, 11, 12, 13, 14, 15);
            s0 = __builtin_amdgcn_wmma_f32_16x16x32_f16(false, kf0, false, qf[cc],
                                                        (short)0, s0, false, false);
            s1 = __builtin_amdgcn_wmma_f32_16x16x32_f16(false, kf1, false, qf[cc],
                                                        (short)0, s1, false, false);
        }

        // ---- Online softmax over these 32 keys (key = v + 8*half, query = l16).
        float mloc = s0[0];
#pragma unroll
        for (int v = 0; v < 8; ++v) {
            mloc = fmaxf(mloc, s0[v]);
            mloc = fmaxf(mloc, s1[v]);
        }
        mloc = fmaxf(mloc, __shfl_xor(mloc, 16, 32));
        const float m_new = fmaxf(m_run, mloc);
        const float alpha = __expf(m_run - m_new);

        v8f p0, p1;
        float ssum = 0.0f;
#pragma unroll
        for (int v = 0; v < 8; ++v) {
            p0[v] = __expf(s0[v] - m_new);
            p1[v] = __expf(s1[v] - m_new);
            ssum += p0[v] + p1[v];
        }
        ssum += __shfl_xor(ssum, 16, 32);
        l_run = l_run * alpha + ssum;
        m_run = m_new;

#pragma unroll
        for (int n = 0; n < 8; ++n)
#pragma unroll
            for (int v = 0; v < 8; ++v) acc[n][v] *= alpha;

        // ---- Repack P (two C-layout frags) -> B-operand frag (32key x 16q).
        v16h pf;
#pragma unroll
        for (int v = 0; v < 8; ++v) {
            float lo, hi;
            if (v < 4) {
                const float t0 = __shfl_xor(p1[2 * v], 16, 32);
                const float t1 = __shfl_xor(p1[2 * v + 1], 16, 32);
                lo = (half_id == 0) ? p0[2 * v]     : t0;
                hi = (half_id == 0) ? p0[2 * v + 1] : t1;
            } else {
                const float t0 = __shfl_xor(p0[2 * v - 8], 16, 32);
                const float t1 = __shfl_xor(p0[2 * v - 7], 16, 32);
                lo = (half_id == 0) ? t0 : p1[2 * v - 8];
                hi = (half_id == 0) ? t1 : p1[2 * v - 7];
            }
            pf[2 * v]     = (half_t)lo;
            pf[2 * v + 1] = (half_t)hi;
        }

        // ---- GEMM2: O += V x P, 8 channel tiles; V A-operand from LDS (2x b128 each).
#pragma unroll
        for (int n = 0; n < 8; ++n) {
            const _Float16* vr = lds_v + (16 * n + l16) * VR_STRIDE;
            const v8h a = *(const v8h*)(vr + 16 * half_id);
            const v8h c = *(const v8h*)(vr + 8 + 16 * half_id);
            const v16h vf = __builtin_shufflevector(a, c, 0, 1, 2, 3, 4, 5, 6, 7,
                                                    8, 9, 10, 11, 12, 13, 14, 15);
            acc[n] = __builtin_amdgcn_wmma_f32_16x16x32_f16(false, vf, false, pf,
                                                            (short)0, acc[n], false, false);
        }
    }

    // ---- Epilogue.
    const float inv_l = 1.0f / l_run;
    float* obase = out + (size_t)b * CH * T_SEQ + qt0 + l16;
#pragma unroll
    for (int n = 0; n < 8; ++n)
#pragma unroll
        for (int v = 0; v < 8; ++v) {
            const int c = 16 * n + v + 8 * half_id;
            obase[(size_t)c * T_SEQ] = acc[n][v] * inv_l;
        }
}

extern "C" void kernel_launch(void* const* d_in, const int* in_sizes, int n_in,
                              void* d_out, int out_size, void* d_ws, size_t ws_size,
                              hipStream_t stream) {
    (void)in_sizes; (void)n_in; (void)d_ws; (void)ws_size; (void)out_size;
    const float* qkv = (const float*)d_in[0];
    float* out = (float*)d_out;
    const int blocks = (NBAT * (T_SEQ / 16)) / WPB;  // 512
    qkv_attn_wmma_lds<<<blocks, 256, 0, stream>>>(qkv, out);
}
